// GraphAttentionLayer_62354335203364
// MI455X (gfx1250) — matrix operand order repaired and checked by reading, not compiled, not run on hardware
//
#include <hip/hip_runtime.h>

#define NN 8192
#define INF_ 256
#define OUTF 128
#define WAVES 8          // K-split waves per workgroup
#define ALPHA 0.2f

typedef __attribute__((ext_vector_type(16))) _Float16 v16h;
typedef __attribute__((ext_vector_type(8)))  _Float16 v8h;
typedef __attribute__((ext_vector_type(8)))  float    v8f;

// ------------------------------------------------- phase 1: Wh = h @ W (f16 WMMA)
// One wave per 16x16 output tile. Stores Wh transposed as f16: WhT[n*NN + i].
__global__ __launch_bounds__(32) void gat_phase1_wh(
    const float* __restrict__ h, const float* __restrict__ W,
    _Float16* __restrict__ WhT) {
  const int tile = blockIdx.x;        // 512 i-tiles * 8 n-tiles
  const int i0 = (tile >> 3) * 16;
  const int nb = (tile & 7) * 16;
  const int l  = threadIdx.x;
  const int r  = l & 15;
  const int hf = l >> 4;
  const int klo = hf * 8;

  v8f c = {0.f,0.f,0.f,0.f,0.f,0.f,0.f,0.f};
  for (int kb = 0; kb < INF_; kb += 32) {
    // A fragment: row i0+r, K = kb + hf*16 + {0..15}
    const float* hrow = h + (size_t)(i0 + r) * INF_ + kb;
    v16h a;
#pragma unroll
    for (int q = 0; q < 8; ++q) a[q]     = (_Float16)hrow[hf * 16 + q];
#pragma unroll
    for (int q = 0; q < 8; ++q) a[8 + q] = (_Float16)hrow[hf * 16 + 8 + q];
    // B fragment: column nb+r, K = kb + hf*16 + {0..15} (strided in W)
    const float* wcol = W + (size_t)(kb + hf * 16) * OUTF + (nb + r);
    v16h b;
#pragma unroll
    for (int q = 0; q < 16; ++q) b[q] = (_Float16)wcol[(size_t)q * OUTF];
    c = __builtin_amdgcn_wmma_f32_16x16x32_f16(false, a, false, b, (short)0, c,
                                               false, false);
  }
  // C layout: VGPR v -> row i0 + v + 8*hf, col nb + r. 8 consecutive rows -> b128 store.
  v8h outv;
#pragma unroll
  for (int v = 0; v < 8; ++v) outv[v] = (_Float16)c[v];
  *(v8h*)(WhT + (size_t)(nb + r) * NN + i0 + 8 * hf) = outv;
}

// ------------------------------------------------- f1 = Wh@a1, f2 = Wh@a2
__global__ void gat_f12(const _Float16* __restrict__ WhT,
                        const float* __restrict__ avec,
                        float* __restrict__ f1, float* __restrict__ f2) {
  const int i = blockIdx.x * blockDim.x + threadIdx.x;  // 0..NN-1
  float s1 = 0.f, s2 = 0.f;
#pragma unroll 4
  for (int n = 0; n < OUTF; ++n) {
    float w = (float)WhT[(size_t)n * NN + i];   // coalesced over i
    s1 = fmaf(w, avec[n], s1);
    s2 = fmaf(w, avec[OUTF + n], s2);
  }
  f1[i] = s1;
  f2[i] = s2;
}

// ------------------------------------------------- fused softmax + att@Wh
__device__ __forceinline__ float gat_s(int adjv, float x) {
  float lr = x > 0.0f ? x : ALPHA * x;      // leaky relu
  float e  = __expf(lr);                    // v_exp_f32
  return adjv > 0 ? e : 0.0f;               // mask (exp(NEG) == 0)
}

// One block (8 waves) per 16-row tile. Wave w owns K-partition w.
// Partials combined via LDS in fixed order (deterministic, no atomics),
// normalized in-block, final tile written straight to out.
__global__ __launch_bounds__(256) void gat_phase2(
    const int* __restrict__ adj, const _Float16* __restrict__ WhT,
    const float* __restrict__ f1, const float* __restrict__ f2,
    float* __restrict__ out) {
  __shared__ float lsum[4 * 16 * OUTF];   // 4 wave slabs (two batches) = 32 KB
  __shared__ float lrs[WAVES * 32];       // per-lane rowsum partials
  __shared__ float lrsF[16];              // final per-row softmax denominators

  const int i0 = blockIdx.x * 16;
  const int tid = threadIdx.x;
  const int w  = tid >> 5;                // wave id == K-split id
  const int l  = tid & 31;
  const int r  = l & 15;
  const int hf = l >> 4;
  const int klo = hf * 8;
  const float f1r = f1[i0 + r];

  const v8f vzero = {0.f,0.f,0.f,0.f,0.f,0.f,0.f,0.f};
  v8f c[8];
#pragma unroll
  for (int t = 0; t < 8; ++t) c[t] = vzero;

  float rs = 0.0f;
  const int  kb0 = w * (NN / WAVES);      // 1024 columns per wave
  const int* adjrow = adj + (size_t)(i0 + r) * NN;

  for (int kc = 0; kc < NN / (32 * WAVES); ++kc) {   // 32 iterations
    const int kb = kb0 + kc * 32;
    __builtin_prefetch(adjrow + kb + 512, 0, 0);     // stream adj ahead

    // 16 adjacency values in the ISA 16-bit A-fragment order
    const int4 j0 = *(const int4*)(adjrow + kb + klo);
    const int4 j1 = *(const int4*)(adjrow + kb + klo + 4);
    const int4 j2 = *(const int4*)(adjrow + kb + 16 + klo);
    const int4 j3 = *(const int4*)(adjrow + kb + 16 + klo + 4);
    const float4 g0 = *(const float4*)(f2 + kb + klo);
    const float4 g1 = *(const float4*)(f2 + kb + klo + 4);
    const float4 g2 = *(const float4*)(f2 + kb + 16 + klo);
    const float4 g3 = *(const float4*)(f2 + kb + 16 + klo + 4);

    float s[16];
    s[0]  = gat_s(j0.x, f1r + g0.x); s[1]  = gat_s(j0.y, f1r + g0.y);
    s[2]  = gat_s(j0.z, f1r + g0.z); s[3]  = gat_s(j0.w, f1r + g0.w);
    s[4]  = gat_s(j1.x, f1r + g1.x); s[5]  = gat_s(j1.y, f1r + g1.y);
    s[6]  = gat_s(j1.z, f1r + g1.z); s[7]  = gat_s(j1.w, f1r + g1.w);
    s[8]  = gat_s(j2.x, f1r + g2.x); s[9]  = gat_s(j2.y, f1r + g2.y);
    s[10] = gat_s(j2.z, f1r + g2.z); s[11] = gat_s(j2.w, f1r + g2.w);
    s[12] = gat_s(j3.x, f1r + g3.x); s[13] = gat_s(j3.y, f1r + g3.y);
    s[14] = gat_s(j3.z, f1r + g3.z); s[15] = gat_s(j3.w, f1r + g3.w);

    v16h a;
#pragma unroll
    for (int q = 0; q < 16; ++q) { rs += s[q]; a[q] = (_Float16)s[q]; }

    // 8 N-tiles reuse the same A fragment; B = WhT two 16B chunks per lane (K = klo.., 16+klo..)
#pragma unroll
    for (int t = 0; t < 8; ++t) {
      const _Float16* bp = WhT + (size_t)(t * 16 + r) * NN + kb;
      const v8h b0 = *(const v8h*)(bp + klo);
      const v8h b1 = *(const v8h*)(bp + 16 + klo);
      const v16h b = __builtin_shufflevector(b0, b1, 0, 1, 2, 3, 4, 5, 6, 7, 8, 9, 10, 11, 12, 13, 14, 15);
      c[t] = __builtin_amdgcn_wmma_f32_16x16x32_f16(false, a, false, b,
                                                    (short)0, c[t], false, false);
    }
  }

  // ---- deterministic cross-wave reduction through LDS (two batches of 4) ----
  lrs[w * 32 + l] = rs;
  if (w < 4) {                       // batch 1: waves 0-3 park their tiles
#pragma unroll
    for (int t = 0; t < 8; ++t)
#pragma unroll
      for (int v = 0; v < 8; ++v)
        lsum[w * 2048 + t * 256 + v * 32 + l] = c[t][v];
  }
  __syncthreads();

  // slot (t=s, v=w, l) read by thread tid -> value for row w+8*hf, col s*16+r
  float acc8[8];
#pragma unroll
  for (int s = 0; s < 8; ++s) {
    const int slot = s * 256 + tid;
    acc8[s] = lsum[slot] + lsum[2048 + slot] + lsum[4096 + slot] +
              lsum[6144 + slot];
  }
  __syncthreads();                   // batch-1 reads complete

  if (w >= 4) {                      // batch 2: waves 4-7
#pragma unroll
    for (int t = 0; t < 8; ++t)
#pragma unroll
      for (int v = 0; v < 8; ++v)
        lsum[(w - 4) * 2048 + t * 256 + v * 32 + l] = c[t][v];
  }
  __syncthreads();

#pragma unroll
  for (int s = 0; s < 8; ++s) {
    const int slot = s * 256 + tid;
    acc8[s] += lsum[slot] + lsum[2048 + slot] + lsum[4096 + slot] +
               lsum[6144 + slot];
  }
  if (tid < 16) {                    // final softmax denominator per row
    float t = 0.0f;
#pragma unroll
    for (int ww = 0; ww < WAVES; ++ww)
      t += lrs[ww * 32 + tid] + lrs[ww * 32 + tid + 16];
    lrsF[tid] = t;
  }
  __syncthreads();

  const int   row  = w + 8 * hf;     // this thread's fixed output row (0..15)
  const float rinv = 1.0f / lrsF[row];
#pragma unroll
  for (int s = 0; s < 8; ++s)
    out[(size_t)(i0 + row) * OUTF + s * 16 + r] = acc8[s] * rinv;
}

// ----------------------------------------------------------------------------
extern "C" void kernel_launch(void* const* d_in, const int* in_sizes, int n_in,
                              void* d_out, int out_size, void* d_ws, size_t ws_size,
                              hipStream_t stream) {
  const float* h    = (const float*)d_in[0];   // (8192,256) f32
  const int*   adj  = (const int*)d_in[1];     // (8192,8192) i32
  const float* W    = (const float*)d_in[2];   // (256,128) f32
  const float* avec = (const float*)d_in[3];   // (256,1) f32
  float* out = (float*)d_out;                  // (8192,128) f32

  char* ws = (char*)d_ws;
  _Float16* WhT = (_Float16*)ws;               // 128*8192*2B = 2 MB
  float* f1     = (float*)(ws + (size_t)2 * 1024 * 1024);
  float* f2     = f1 + NN;                     // total ~2.1 MB of ws

  gat_phase1_wh<<<dim3((NN / 16) * (OUTF / 16)), 32, 0, stream>>>(h, W, WhT);
  gat_f12<<<dim3(NN / 256), 256, 0, stream>>>(WhT, avec, f1, f2);
  gat_phase2<<<dim3(NN / 16), 256, 0, stream>>>(adj, WhT, f1, f2, out);
}
